// QMixtralSparseMoeBlock_59554016526379
// MI455X (gfx1250) — compile-verified, matrix operand order checked by reading, not run on hardware
//
#include <hip/hip_runtime.h>
#include <hip/hip_bf16.h>

typedef __attribute__((ext_vector_type(16))) __bf16 v16bf;
typedef __attribute__((ext_vector_type(8)))  float  v8f;

constexpr int Tn = 1024;   // B*S tokens
constexpr int Hn = 1024;   // hidden
constexpr int Fn = 3584;   // ffn
constexpr int En = 8;      // experts
constexpr int MT = 64;     // token tile (4 WMMA M-tiles)
constexpr int MAX_TILES = Tn / MT;   // 16: worst case one expert owns every token
constexpr int NF_BLK = 128;          // f-columns per block (8 waves * 16)
constexpr int NH_BLK = 128;          // h-columns per block (8 waves * 16)
constexpr int KSPLIT = 2;            // down-proj F split (partials via atomic add)
constexpr int FS = Fn / KSPLIT;      // 1792 (56 K-iters per chunk)

// ---- workspace layout (bytes) ----
constexpr size_t XB_OFF  = 0;                              // x in bf16: T*H u16  (2 MB)
constexpr size_t G_OFF   = XB_OFF + (size_t)Tn*Hn*2;       // gated act: (2T)*F u16 (14 MB)
constexpr size_t DW_OFF  = G_OFF + (size_t)(2*Tn)*Fn*2;    // dense routing w: T*E f32
constexpr size_t IDS_OFF = DW_OFF + (size_t)Tn*En*4;       // per-expert token ids: E*T i32
constexpr size_t CNT_OFF = IDS_OFF + (size_t)En*Tn*4;      // counts: E i32
constexpr size_t BAS_OFF = CNT_OFF + (size_t)En*4;         // bases:  E i32

__device__ __forceinline__ unsigned short f2bf(float f) {
  return __builtin_bit_cast(unsigned short, (__bf16)f);   // hardware RNE convert
}

__device__ __forceinline__ float fast_rcp(float x) {
#if __has_builtin(__builtin_amdgcn_rcpf)
  return __builtin_amdgcn_rcpf(x);    // v_rcp_f32, ~1 ulp; output feeds bf16 anyway
#else
  return 1.f / x;
#endif
}

// load 16 consecutive fp32, convert to a bf16 WMMA fragment half-row
__device__ __forceinline__ v16bf cvt16(const float* __restrict__ p) {
  const float4* q = (const float4*)p;
  v16bf v;
#pragma unroll
  for (int i = 0; i < 4; ++i) {
    float4 f = q[i];
    v[4*i+0] = (__bf16)f.x;
    v[4*i+1] = (__bf16)f.y;
    v[4*i+2] = (__bf16)f.z;
    v[4*i+3] = (__bf16)f.w;
  }
  return v;
}

__device__ __forceinline__ v16bf ld16(const unsigned short* __restrict__ p) {
  return *(const v16bf*)p;   // 32B contiguous, 32B aligned
}

__device__ __forceinline__ v8f wmma_bf16(v16bf a, v16bf b, v8f c) {
  return __builtin_amdgcn_wmma_f32_16x16x32_bf16(false, a, false, b, (short)0, c,
                                                 false, false);
}

__device__ __forceinline__ void atomicAddF(float* p, float v) {
  __hip_atomic_fetch_add(p, v, __ATOMIC_RELAXED, __HIP_MEMORY_SCOPE_AGENT);
}

// ---------------- kernel 0: zero out + counters ----------------
__global__ __launch_bounds__(256) void k_init(float* __restrict__ out,
                                              int* __restrict__ cnt) {
  int i = blockIdx.x * blockDim.x + threadIdx.x;   // 262144 threads, 4 floats each
  float4 z = {0.f, 0.f, 0.f, 0.f};
  ((float4*)out)[i] = z;
  if (i < En) cnt[i] = 0;
}

// ---------------- kernel 1: router + bf16 convert + gather lists ----------------
__global__ __launch_bounds__(256) void k_router(
    const float* __restrict__ x, const float* __restrict__ gw,
    float* __restrict__ logits, float* __restrict__ dw,
    unsigned short* __restrict__ xb, int* __restrict__ cnt, int* __restrict__ ids) {
  const int t = blockIdx.x;
  const int lane = threadIdx.x & 31;
  const int e = threadIdx.x >> 5;                  // wave -> expert (wave32)
  const float* xr = x + (size_t)t * Hn;
  const float* gr = gw + (size_t)e * Hn;

  float s = 0.f;
  for (int i = lane; i < Hn; i += 32) s += xr[i] * gr[i];
#pragma unroll
  for (int off = 16; off; off >>= 1) s += __shfl_xor(s, off);

  __shared__ float lg[En];
  if (lane == 0) lg[e] = s;

  // fp32 -> bf16 conversion of this token's row
  for (int i = threadIdx.x; i < Hn; i += 256) xb[(size_t)t * Hn + i] = f2bf(xr[i]);
  __syncthreads();

  if (threadIdx.x == 0) {
    float l[En], m = -3.4e38f;
#pragma unroll
    for (int j = 0; j < En; ++j) { l[j] = lg[j]; logits[t * En + j] = l[j]; m = fmaxf(m, l[j]); }
    float p[En];
#pragma unroll
    for (int j = 0; j < En; ++j) p[j] = __expf(l[j] - m);
    int i0 = 0;
#pragma unroll
    for (int j = 1; j < En; ++j) if (p[j] > p[i0]) i0 = j;     // first-max tie-break
    int i1 = (i0 == 0) ? 1 : 0;
#pragma unroll
    for (int j = 0; j < En; ++j) if (j != i0 && p[j] > p[i1]) i1 = j;
    float inv = 1.f / (p[i0] + p[i1]);
    float w0 = p[i0] * inv, w1v = p[i1] * inv;
#pragma unroll
    for (int j = 0; j < En; ++j)
      dw[t * En + j] = (j == i0) ? w0 : ((j == i1) ? w1v : 0.f);
    int p0 = atomicAdd(&cnt[i0], 1); ids[i0 * Tn + p0] = t;
    int p1 = atomicAdd(&cnt[i1], 1); ids[i1 * Tn + p1] = t;
  }
}

// ---------------- kernel 2: tiny exclusive scan ----------------
__global__ void k_scan(const int* __restrict__ cnt, int* __restrict__ bas) {
  int acc = 0;
  for (int e = 0; e < En; ++e) { bas[e] = acc; acc += cnt[e]; }
}

// ---------------- kernel 3: gate/up projections + SwiGLU ----------------
// grid (MAX_TILES, Fn/NF_BLK, En), block 256 (8 waves, 16 f-cols each)
__global__ __launch_bounds__(256) void k_moe_up(
    const unsigned short* __restrict__ xb,
    const float* __restrict__ w1, const float* __restrict__ w3,
    const int* __restrict__ ids, const int* __restrict__ cnt,
    const int* __restrict__ bas, unsigned short* __restrict__ g) {
  const int e = blockIdx.z;
  const int c = cnt[e];
  const int row0 = blockIdx.x * MT;
  if (row0 >= c) return;

  const int wave = threadIdx.x >> 5, lane = threadIdx.x & 31;
  const int n = lane & 15;                 // A: row id; B: col id
  const int kOff = (lane >> 4) * 16;       // K half selector
  const int f0 = blockIdx.y * NF_BLK + wave * 16;
  const int* el = ids + e * Tn;

  const unsigned short* xr[4];
#pragma unroll
  for (int m = 0; m < 4; ++m) {
    int tid = el[min(row0 + 16 * m + n, c - 1)];
    xr[m] = xb + (size_t)tid * Hn + kOff;
  }
  const float* w1p = w1 + ((size_t)e * Fn + f0 + n) * Hn + kOff;
  const float* w3p = w3 + ((size_t)e * Fn + f0 + n) * Hn + kOff;

  v8f acc1[4], acc3[4];
#pragma unroll
  for (int m = 0; m < 4; ++m) {
    acc1[m] = (v8f){0,0,0,0,0,0,0,0};
    acc3[m] = (v8f){0,0,0,0,0,0,0,0};
  }

  for (int k = 0; k < Hn; k += 32) {
    v16bf b1 = cvt16(w1p + k);
    v16bf b3 = cvt16(w3p + k);
#pragma unroll
    for (int m = 0; m < 4; ++m) {
      v16bf a = ld16(xr[m] + k);
      acc1[m] = wmma_bf16(a, b1, acc1[m]);
      acc3[m] = wmma_bf16(a, b3, acc3[m]);
    }
  }

  const int gb = bas[e];
  const int mBase = (lane >> 4) * 8;       // C/D layout: lanes 16-31 hold M=8..15
#pragma unroll
  for (int m = 0; m < 4; ++m) {
#pragma unroll
    for (int r = 0; r < 8; ++r) {
      int ri = row0 + 16 * m + mBase + r;
      if (ri < c) {
        float a = acc1[m][r];
        // silu(a)*h3 = a * sigmoid(a) * h3 ; fast rcp is fine (result -> bf16)
        float v = a * fast_rcp(1.f + __expf(-a)) * acc3[m][r];
        g[(size_t)(gb + ri) * Fn + f0 + n] = f2bf(v);
      }
    }
  }
}

// ---------------- kernel 4: down projection + weighted combine ----------------
// grid (MAX_TILES, (Hn/NH_BLK)*KSPLIT, En), block 256 (8 waves, 16 h-cols each)
__global__ __launch_bounds__(256) void k_moe_down(
    const unsigned short* __restrict__ g, const float* __restrict__ w2,
    const int* __restrict__ ids, const int* __restrict__ cnt,
    const int* __restrict__ bas, const float* __restrict__ dw,
    float* __restrict__ out) {
  const int e = blockIdx.z;
  const int c = cnt[e];
  const int row0 = blockIdx.x * MT;
  if (row0 >= c) return;

  const int wave = threadIdx.x >> 5, lane = threadIdx.x & 31;
  const int n = lane & 15;
  const int kOff = (lane >> 4) * 16;
  const int hgrp = blockIdx.y % (Hn / NH_BLK);
  const int kchk = blockIdx.y / (Hn / NH_BLK);
  const int h0 = hgrp * NH_BLK + wave * 16;
  const int k0 = kchk * FS;
  const int gb = bas[e];
  const int* el = ids + e * Tn;

  const unsigned short* gp[4];
#pragma unroll
  for (int m = 0; m < 4; ++m)
    gp[m] = g + (size_t)(gb + min(row0 + 16 * m + n, c - 1)) * Fn + kOff + k0;
  const float* w2p = w2 + ((size_t)e * Hn + h0 + n) * Fn + kOff + k0;

  v8f acc[4];
#pragma unroll
  for (int m = 0; m < 4; ++m) acc[m] = (v8f){0,0,0,0,0,0,0,0};

  for (int k = 0; k < FS; k += 32) {
    v16bf b = cvt16(w2p + k);
#pragma unroll
    for (int m = 0; m < 4; ++m) {
      v16bf a = ld16(gp[m] + k);
      acc[m] = wmma_bf16(a, b, acc[m]);
    }
  }

  const int mBase = (lane >> 4) * 8;
#pragma unroll
  for (int m = 0; m < 4; ++m) {
#pragma unroll
    for (int r = 0; r < 8; ++r) {
      int ri = row0 + 16 * m + mBase + r;
      if (ri < c) {
        int t = el[ri];
        float w = dw[t * En + e];
        atomicAddF(out + (size_t)t * Hn + h0 + n, acc[m][r] * w);
      }
    }
  }
}

extern "C" void kernel_launch(void* const* d_in, const int* in_sizes, int n_in,
                              void* d_out, int out_size, void* d_ws, size_t ws_size,
                              hipStream_t stream) {
  const float* x  = (const float*)d_in[0];   // [2,512,1024]
  const float* gw = (const float*)d_in[1];   // [8,1024]
  const float* w1 = (const float*)d_in[2];   // [8,3584,1024]
  const float* w2 = (const float*)d_in[3];   // [8,1024,3584]
  const float* w3 = (const float*)d_in[4];   // [8,3584,1024]

  float* out    = (float*)d_out;             // [T,H]
  float* logits = out + (size_t)Tn * Hn;     // [T,E]

  char* ws = (char*)d_ws;
  unsigned short* xb = (unsigned short*)(ws + XB_OFF);
  unsigned short* g  = (unsigned short*)(ws + G_OFF);
  float* dw  = (float*)(ws + DW_OFF);
  int*   ids = (int*)(ws + IDS_OFF);
  int*   cnt = (int*)(ws + CNT_OFF);
  int*   bas = (int*)(ws + BAS_OFF);

  k_init<<<dim3((Tn * Hn / 4) / 256), dim3(256), 0, stream>>>(out, cnt);
  k_router<<<dim3(Tn), dim3(256), 0, stream>>>(x, gw, logits, dw, xb, cnt, ids);
  k_scan<<<dim3(1), dim3(1), 0, stream>>>(cnt, bas);
  k_moe_up<<<dim3(MAX_TILES, Fn / NF_BLK, En), dim3(256), 0, stream>>>(
      xb, w1, w3, ids, cnt, bas, g);
  k_moe_down<<<dim3(MAX_TILES, (Hn / NH_BLK) * KSPLIT, En), dim3(256), 0, stream>>>(
      g, w2, ids, cnt, bas, dw, out);
}
